// RF_Pool_80178449482040
// MI455X (gfx1250) — compile-verified, matrix operand order checked by reading, not run on hardware
//
#include <hip/hip_runtime.h>

typedef __attribute__((ext_vector_type(16))) _Float16 v16h;
typedef __attribute__((ext_vector_type(8)))  float    v8f;

#define HW      3136
#define KTOT    196
#define NT      256
#define NWAVES  8
#define NTILES  98          // HW / 32

__global__ __launch_bounds__(NT)
void rfpool_prob_kernel(const float* __restrict__ u,
                        const float* __restrict__ rfs,
                        float* __restrict__ out)
{
    __shared__ float    u_lds[HW];     // one (b,c) activation row, 12.25 KB
    __shared__ unsigned mmax[16];      // per-k running max (nonneg -> uint-ordered)
    __shared__ float    ssum[16];      // per-k partial sums across waves

    const int tid  = threadIdx.x;
    const int lane = tid & 31;
    const int wave = tid >> 5;
    const int bc   = blockIdx.y;        // 0 .. B*C-1
    const int k0   = blockIdx.x << 4;   // k-tile base

    // WMMA f16 A-matrix lane mapping: lanes 0-15 -> rows, K-halves {0..7,16..23};
    // lanes 16-31 -> same rows, K-halves {8..15,24..31}.
    const int krow = lane & 15;
    const int half = lane >> 4;
    const int base = half << 3;

    int kk = k0 + krow;                 // clamp last partial k-tile in-bounds
    if (kk > KTOT - 1) kk = KTOT - 1;

    // ---- stage u[b,c,:] into LDS ----
    const float* urow = u + (size_t)bc * HW;
    for (int i = tid; i < HW; i += NT) u_lds[i] = urow[i];
    if (tid < 16) { mmax[tid] = 0u; ssum[tid] = 0.0f; }
    __syncthreads();

    // ---- phase 1: m_k = max(0, max_hw u*rf) ----
    float lmax = 0.0f;                  // implicit off-unit logit 0
    for (int t = wave; t < NTILES; t += NWAVES) {
        const int h0 = t << 5;
        const float* rp = rfs + (size_t)h0 * KTOT + kk;
        if (t + NWAVES < NTILES)        // warm next tile of rfs for this wave
            __builtin_prefetch(rp + (size_t)(NWAVES << 5) * KTOT, 0, 3);
        #pragma unroll
        for (int e = 0; e < 16; ++e) {
            const int off = base + (e & 7) + ((e >> 3) << 4);
            const float l = u_lds[h0 + off] * rp[(size_t)off * KTOT];
            lmax = fmaxf(lmax, l);
        }
    }
    atomicMax(&mmax[krow], __float_as_uint(lmax));   // ds_max_u32
    __syncthreads();

    const float m = __uint_as_float(mmax[krow]);

    // ---- phase 2: S_k = sum_hw exp(u*rf - m) via v_wmma_f32_16x16x32_f16 ----
    v16h ones;
    #pragma unroll
    for (int e = 0; e < 16; ++e) ones[e] = (_Float16)1.0f;  // B = ones: D cols = row sums
    v8f acc = {};

    for (int t = wave; t < NTILES; t += NWAVES) {
        const int h0 = t << 5;
        const float* rp = rfs + (size_t)h0 * KTOT + kk;
        v16h a;
        #pragma unroll
        for (int e = 0; e < 16; ++e) {
            const int off = base + (e & 7) + ((e >> 3) << 4);
            const float l = u_lds[h0 + off] * rp[(size_t)off * KTOT];
            a[e] = (_Float16)__expf(l - m);          // e in [0,1]
        }
        acc = __builtin_amdgcn_wmma_f32_16x16x32_f16(
                  false, a, false, ones, (short)0, acc, false, false);
    }

    // C/D layout: VGPR i = row i (lanes 0-15) or row 8+i (lanes 16-31);
    // all N columns identical, so lanes 0 and 16 carry the 16 row sums.
    if (krow == 0) {
        #pragma unroll
        for (int i = 0; i < 8; ++i)
            atomicAdd(&ssum[base + i], acc[i]);      // ds_add_f32
    }
    __syncthreads();

    // ---- finalize: p = S / (S + exp(-m)) ----
    if (tid < 16) {
        const int k = k0 + tid;
        if (k < KTOT) {
            const float mm = __uint_as_float(mmax[tid]);
            const float S  = ssum[tid];
            out[(size_t)bc * KTOT + k] = S / (S + __expf(-mm));
        }
    }
}

extern "C" void kernel_launch(void* const* d_in, const int* in_sizes, int n_in,
                              void* d_out, int out_size, void* d_ws, size_t ws_size,
                              hipStream_t stream)
{
    (void)in_sizes; (void)n_in; (void)d_ws; (void)ws_size; (void)out_size;
    const float* u   = (const float*)d_in[0];   // (8,32,56,56) f32
    const float* rfs = (const float*)d_in[1];   // (56,56,196)  f32
    float* out       = (float*)d_out;           // (8,32,196)   f32

    dim3 grid((KTOT + 15) / 16, 8 * 32);        // 13 k-tiles x 256 (b,c)
    rfpool_prob_kernel<<<grid, NT, 0, stream>>>(u, rfs, out);
}
